// GatedUpdate_5076651344236
// MI455X (gfx1250) — matrix operand order repaired and checked by reading, not compiled, not run on hardware
//
#include <hip/hip_runtime.h>
#include <hip/hip_bf16.h>
#include <math.h>

// HGT-GNN forward for MI455X (gfx1250): bf16 WMMA GEMMs + atomic segment-softmax.
#define HC 128
#define TILE_ROWS 64

typedef __attribute__((ext_vector_type(16))) __bf16 v16bf;
typedef __attribute__((ext_vector_type(8)))  float  v8f;

__device__ __forceinline__ float gelu_exact(float x) {
    return 0.5f * x * (1.0f + erff(x * 0.7071067811865476f));
}

__device__ __forceinline__ void atomic_max_f32(float* addr, float val) {
    unsigned int* ai = (unsigned int*)addr;
    unsigned int old = __float_as_uint(*(volatile float*)addr);
    while (__uint_as_float(old) < val) {
        unsigned int assumed = old;
        old = atomicCAS(ai, assumed, __float_as_uint(val));
        if (old == assumed) break;
    }
}

// ---------------------------------------------------------------------------
// C[N,128] = actout( actin(A)[N,128] @ W[128,128] + bias ), optional gated skip
// ACT_IN: 0=none 1=gelu(exact)   ACT_OUT: 0=none 1=relu
// GATED : new = g*out + (1-g)*res,  g = sigmoid(*skip_ptr)
// LDS plan: [sA 64x132 bf16 | sW 128x132 bf16] during compute, then the whole
// block re-aliased as float sC[64][128] for the coalesced store phase.
// ---------------------------------------------------------------------------
template<int ACT_IN, int ACT_OUT, bool GATED>
__global__ __launch_bounds__(128)
void gemm128_kernel(const float* __restrict__ A, const float* __restrict__ W,
                    const float* __restrict__ bias, float* __restrict__ C, int N,
                    const float* __restrict__ res, const float* __restrict__ skip_ptr) {
    __shared__ __attribute__((aligned(16))) unsigned char smem[50688];
    __bf16 (*sA)[132] = reinterpret_cast<__bf16(*)[132]>(smem);
    __bf16 (*sW)[132] = reinterpret_cast<__bf16(*)[132]>(smem + TILE_ROWS * 132 * 2);
    float*  sC        = reinterpret_cast<float*>(smem);

    const int tid  = threadIdx.x;
    const int row0 = blockIdx.x * TILE_ROWS;
    const bool full = (row0 + TILE_ROWS) <= N;

    // stage W (transposed) into LDS as bf16 — float4 coalesced global reads
    const float4* W4 = (const float4*)W;
    for (int i4 = tid; i4 < HC * (HC / 4); i4 += 128) {
        int k = i4 >> 5, c = (i4 & 31) * 4;
        float4 w4 = W4[i4];
        sW[c + 0][k] = (__bf16)w4.x;
        sW[c + 1][k] = (__bf16)w4.y;
        sW[c + 2][k] = (__bf16)w4.z;
        sW[c + 3][k] = (__bf16)w4.w;
    }
    // stage A tile into LDS as bf16 (+ optional input activation)
    const float4* A4 = (const float4*)(A + (size_t)row0 * HC);
    for (int i4 = tid; i4 < TILE_ROWS * (HC / 4); i4 += 128) {
        int r = i4 >> 5, c = (i4 & 31) * 4;
        float4 x4;
        if (full || (row0 + r) < N) x4 = A4[i4];
        else                        x4 = make_float4(0.f, 0.f, 0.f, 0.f);
        if (ACT_IN == 1) {
            x4.x = gelu_exact(x4.x); x4.y = gelu_exact(x4.y);
            x4.z = gelu_exact(x4.z); x4.w = gelu_exact(x4.w);
        }
        __bf16* p = &sA[r][c];
        p[0] = (__bf16)x4.x; p[1] = (__bf16)x4.y;
        p[2] = (__bf16)x4.z; p[3] = (__bf16)x4.w;
    }
    __syncthreads();

    const int wave = tid >> 5;
    const int lane = tid & 31;
    const int half = lane >> 4;     // ISA: lanes 16-31 hold the "upper" K/M slice
    const int m    = lane & 15;
    const int mrow = wave * 16;     // this wave's 16-row block within the tile

    v8f acc[8] = {};

    #pragma unroll
    for (int kk = 0; kk < 4; ++kk) {                    // K = 128 = 4 x 32
        // A fragment 16x32 bf16: elem j -> K = half*16 + j
        v16bf afrag;
        const __bf16* ar = &sA[mrow + m][kk * 32];
        #pragma unroll
        for (int j = 0; j < 8; ++j) {
            afrag[j]     = ar[half * 16 + j];
            afrag[8 + j] = ar[half * 16 + 8 + j];
        }
        #pragma unroll
        for (int ct = 0; ct < 8; ++ct) {                // 8 column tiles of 16
            // B fragment 32x16: N = lane&15, elem j -> K = half*16 + j
            v16bf bfrag;
            const __bf16* br = &sW[ct * 16 + m][kk * 32 + half * 16];
            #pragma unroll
            for (int j = 0; j < 16; ++j) bfrag[j] = br[j];
            acc[ct] = __builtin_amdgcn_wmma_f32_16x16x32_bf16(
                false, afrag, false, bfrag, (short)0, acc[ct], false, false);
        }
    }
    __syncthreads();   // everyone done reading sA/sW; re-alias as float sC

    // scatter accumulators into LDS (C/D layout: vgpr r -> row half*8+r, col lane&15)
    #pragma unroll
    for (int ct = 0; ct < 8; ++ct) {
        #pragma unroll
        for (int r = 0; r < 8; ++r)
            sC[(mrow + half * 8 + r) * HC + ct * 16 + m] = acc[ct][r];
    }
    __syncthreads();

    // coalesced float4 epilogue + store
    float g = 0.0f;
    if (GATED) g = 1.0f / (1.0f + __expf(-skip_ptr[0]));
    const float4* bias4 = (const float4*)bias;
    const float4* sC4   = (const float4*)sC;
    for (int i4 = tid; i4 < TILE_ROWS * (HC / 4); i4 += 128) {
        int r = i4 >> 5, c4 = i4 & 31;
        int row = row0 + r;
        if (!full && row >= N) continue;
        float4 o = sC4[i4];
        float4 b = bias4[c4];
        o.x += b.x; o.y += b.y; o.z += b.z; o.w += b.w;
        if (ACT_OUT == 1) {
            o.x = fmaxf(o.x, 0.f); o.y = fmaxf(o.y, 0.f);
            o.z = fmaxf(o.z, 0.f); o.w = fmaxf(o.w, 0.f);
        }
        if (GATED) {
            float4 rv = ((const float4*)res)[(size_t)row * (HC / 4) + c4];
            float h = 1.0f - g;
            o.x = g * o.x + h * rv.x; o.y = g * o.y + h * rv.y;
            o.z = g * o.z + h * rv.z; o.w = g * o.w + h * rv.w;
        }
        ((float4*)C)[(size_t)row * (HC / 4) + c4] = o;
    }
}

// ---------------------------------------------------------------------------
// Relation transform: C[n, h*32+e] = sum_d A[n, h*32+d] * R[h][d][e]
// (block-diagonal GEMM: one WMMA per 16-col tile, K limited to the head's 32)
// ---------------------------------------------------------------------------
__global__ __launch_bounds__(128)
void rel_transform_kernel(const float* __restrict__ A, const float* __restrict__ R,
                          float* __restrict__ C, int N) {
    __shared__ __attribute__((aligned(16))) unsigned char smem[32768];
    __bf16 (*sA)[132]     = reinterpret_cast<__bf16(*)[132]>(smem);
    __bf16 (*sRt)[32][33] = reinterpret_cast<__bf16(*)[32][33]>(smem + TILE_ROWS * 132 * 2);
    float*  sC            = reinterpret_cast<float*>(smem);

    const int tid  = threadIdx.x;
    const int row0 = blockIdx.x * TILE_ROWS;
    const bool full = (row0 + TILE_ROWS) <= N;

    // stage R transposed: sRt[h][e][d]   (R flat: h*1024 + d*32 + e)
    const float4* R4 = (const float4*)R;
    for (int i4 = tid; i4 < 1024; i4 += 128) {
        int h = i4 >> 8, d = (i4 >> 3) & 31, e = (i4 & 7) * 4;
        float4 r4 = R4[i4];
        sRt[h][e + 0][d] = (__bf16)r4.x;
        sRt[h][e + 1][d] = (__bf16)r4.y;
        sRt[h][e + 2][d] = (__bf16)r4.z;
        sRt[h][e + 3][d] = (__bf16)r4.w;
    }
    const float4* A4 = (const float4*)(A + (size_t)row0 * HC);
    for (int i4 = tid; i4 < TILE_ROWS * (HC / 4); i4 += 128) {
        int r = i4 >> 5, c = (i4 & 31) * 4;
        float4 x4;
        if (full || (row0 + r) < N) x4 = A4[i4];
        else                        x4 = make_float4(0.f, 0.f, 0.f, 0.f);
        __bf16* p = &sA[r][c];
        p[0] = (__bf16)x4.x; p[1] = (__bf16)x4.y;
        p[2] = (__bf16)x4.z; p[3] = (__bf16)x4.w;
    }
    __syncthreads();

    const int wave = tid >> 5;
    const int lane = tid & 31;
    const int half = lane >> 4;
    const int m    = lane & 15;
    const int mrow = wave * 16;

    v8f acc[8];
    #pragma unroll
    for (int ct = 0; ct < 8; ++ct) {
        int head = ct >> 1;
        int nh   = (ct & 1) * 16;           // which 16-col half of the head
        v16bf afrag, bfrag;
        const __bf16* ar = &sA[mrow + m][head * 32];
        #pragma unroll
        for (int j = 0; j < 8; ++j) {
            afrag[j]     = ar[half * 16 + j];
            afrag[8 + j] = ar[half * 16 + 8 + j];
        }
        const __bf16* br = &sRt[head][nh + m][half * 16];
        #pragma unroll
        for (int j = 0; j < 16; ++j) bfrag[j] = br[j];
        v8f z = {};
        acc[ct] = __builtin_amdgcn_wmma_f32_16x16x32_bf16(
            false, afrag, false, bfrag, (short)0, z, false, false);
    }
    __syncthreads();

    #pragma unroll
    for (int ct = 0; ct < 8; ++ct) {
        #pragma unroll
        for (int r = 0; r < 8; ++r)
            sC[(mrow + half * 8 + r) * HC + ct * 16 + m] = acc[ct][r];
    }
    __syncthreads();

    const float4* sC4 = (const float4*)sC;
    for (int i4 = tid; i4 < TILE_ROWS * (HC / 4); i4 += 128) {
        int r = i4 >> 5, c4 = i4 & 31;
        int row = row0 + r;
        if (!full && row >= N) continue;
        ((float4*)C)[(size_t)row * (HC / 4) + c4] = sC4[i4];
    }
}

// ---------------------------------------------------------------------------
// Edge attention (segment softmax) in three passes
// ---------------------------------------------------------------------------
__global__ void attn_score_kernel(const int* __restrict__ src, const int* __restrict__ dst,
                                  const float* __restrict__ q, const float* __restrict__ kt,
                                  const float* __restrict__ p_rel, float* __restrict__ alpha,
                                  float* __restrict__ amax, int E) {
    int idx = blockIdx.x * blockDim.x + threadIdx.x;
    if (idx >= E * 4) return;
    int e = idx >> 2, h = idx & 3;
    int s = src[e], d = dst[e];
    const float4* qp = (const float4*)&q[(size_t)d * HC + h * 32];
    const float4* kp = (const float4*)&kt[(size_t)s * HC + h * 32];
    float acc = 0.0f;
    #pragma unroll
    for (int i = 0; i < 8; ++i) {
        float4 a4 = qp[i], b4 = kp[i];
        acc += a4.x * b4.x + a4.y * b4.y + a4.z * b4.z + a4.w * b4.w;
    }
    float a = acc * p_rel[h] * 0.17677669529663688f;   // 1/sqrt(32)
    alpha[idx] = a;
    atomic_max_f32(&amax[d * 4 + h], a);
}

__global__ void attn_norm_kernel(const int* __restrict__ dst, float* __restrict__ alpha,
                                 const float* __restrict__ amax, float* __restrict__ den, int E) {
    int idx = blockIdx.x * blockDim.x + threadIdx.x;
    if (idx >= E * 4) return;
    int e = idx >> 2, h = idx & 3;
    int d = dst[e];
    float ex = __expf(alpha[idx] - amax[d * 4 + h]);
    alpha[idx] = ex;                                    // alpha now holds exp()
    atomicAdd(&den[d * 4 + h], ex);
}

__global__ void attn_scatter_kernel(const int* __restrict__ src, const int* __restrict__ dst,
                                    const float* __restrict__ vt, const float* __restrict__ ex,
                                    const float* __restrict__ den, float* __restrict__ out, int E) {
    int idx = blockIdx.x * blockDim.x + threadIdx.x;
    if (idx >= E * HC) return;
    int e = idx >> 7, c = idx & 127, h = c >> 5;
    int s = src[e], d = dst[e];
    float w = ex[e * 4 + h] / den[d * 4 + h];
    atomicAdd(&out[(size_t)d * HC + c], vt[(size_t)s * HC + c] * w);
}

__global__ void fill_kernel(float* __restrict__ p, float v, int n) {
    int i = blockIdx.x * blockDim.x + threadIdx.x;
    if (i < n) p[i] = v;
}

// output head: logits = x@w2 + b2 (128->2), softmax, atomic pooled sums
__global__ void head_pool_kernel(const float* __restrict__ x, const float* __restrict__ w2,
                                 const float* __restrict__ b2, const int* __restrict__ batch,
                                 float* __restrict__ pooled, int N, int G) {
    int n = blockIdx.x * blockDim.x + threadIdx.x;
    if (n >= N) return;
    float l0 = b2[0], l1 = b2[1];
    const float4* xr = (const float4*)&x[(size_t)n * HC];
    const float4* w4 = (const float4*)w2;   // [128,2] -> pairs of k per float4
    #pragma unroll 8
    for (int k4 = 0; k4 < HC / 4; ++k4) {
        float4 xv = xr[k4];
        float4 f0 = w4[k4 * 2], f1 = w4[k4 * 2 + 1];
        l0 += xv.x * f0.x + xv.y * f0.z + xv.z * f1.x + xv.w * f1.z;
        l1 += xv.x * f0.y + xv.y * f0.w + xv.z * f1.y + xv.w * f1.w;
    }
    float mx = fmaxf(l0, l1);
    float p0 = __expf(l0 - mx), p1 = __expf(l1 - mx);
    float s = p0 + p1;
    p0 /= s; p1 /= s;
    int g = batch[n];
    atomicAdd(&pooled[g * 2],     p0);
    atomicAdd(&pooled[g * 2 + 1], p1);
    atomicAdd(&pooled[G * 2 + g], 1.0f);
}

__global__ void finalize_kernel(const float* __restrict__ pooled, float* __restrict__ out, int G) {
    int i = blockIdx.x * blockDim.x + threadIdx.x;
    if (i >= G * 2) return;
    int g = i >> 1;
    out[i] = pooled[i] / fmaxf(pooled[G * 2 + g], 1.0f);
}

// ---------------------------------------------------------------------------
extern "C" void kernel_launch(void* const* d_in, const int* in_sizes, int n_in,
                              void* d_out, int out_size, void* d_ws, size_t ws_size,
                              hipStream_t stream) {
    (void)n_in; (void)ws_size;
    const float* x_var   = (const float*)d_in[0];
    const float* x_con   = (const float*)d_in[1];
    const float* in_w    = (const float*)d_in[2];   // [2,3,128,128]
    const float* in_b    = (const float*)d_in[3];   // [2,3,128]
    const float* out_w0  = (const float*)d_in[4];
    const float* out_b0  = (const float*)d_in[5];
    const float* out_w1  = (const float*)d_in[6];
    const float* out_b1  = (const float*)d_in[7];
    const float* out_w2  = (const float*)d_in[8];
    const float* out_b2  = (const float*)d_in[9];
    const float* k_w     = (const float*)d_in[10];  // [L,2,128,128]
    const float* k_b     = (const float*)d_in[11];
    const float* q_w     = (const float*)d_in[12];
    const float* q_b     = (const float*)d_in[13];
    const float* v_w     = (const float*)d_in[14];
    const float* v_b     = (const float*)d_in[15];
    const float* a_w     = (const float*)d_in[16];
    const float* a_b     = (const float*)d_in[17];
    const float* skip    = (const float*)d_in[18];  // [L,2]
    const float* a_rel   = (const float*)d_in[19];  // [L,2,4,32,32]
    const float* m_rel   = (const float*)d_in[20];
    const float* p_rel   = (const float*)d_in[21];  // [L,2,4]
    const int*   evc     = (const int*)d_in[22];    // [2,E]
    const int*   ecv     = (const int*)d_in[23];
    const int*   batch   = (const int*)d_in[24];

    const int Nv = in_sizes[0] / HC;
    const int Nc = in_sizes[1] / HC;
    const int E  = in_sizes[22] / 2;
    const int L  = in_sizes[18] / 2;
    const int G  = out_size / 2;
    const int Nmax = (Nv > Nc) ? Nv : Nc;

    // workspace carve (floats)
    float* ws = (float*)d_ws;
    const size_t U = (size_t)Nmax * HC;
    float* X0a  = ws + 0 * U;
    float* X0b  = ws + 1 * U;
    float* X1a  = ws + 2 * U;
    float* X1b  = ws + 3 * U;
    float* T0   = ws + 4 * U;
    float* KT   = ws + 5 * U;   // also second MLP ping-pong buffer
    float* VT   = ws + 6 * U;
    float* Q    = ws + 7 * U;
    float* AGGC = ws + 8 * U;
    float* AGGV = ws + 9 * U;
    float* alpha  = ws + 10 * U;             // [E,4]
    float* amax   = alpha + (size_t)E * 4;   // [Nmax,4]
    float* den    = amax + (size_t)Nmax * 4; // [Nmax,4]
    float* pooled = den + (size_t)Nmax * 4;  // [G*2 sums][G counts]

    const dim3 TB(128);
    auto gblk = [](int n) { return dim3((unsigned)((n + TILE_ROWS - 1) / TILE_ROWS)); };
    #define FILL(ptr, val, n) fill_kernel<<<dim3((unsigned)(((n) + 255) / 256)), dim3(256), 0, stream>>>(ptr, val, n)

    // ---- input MLPs (relu, relu, none) ----
    gemm128_kernel<0,1,false><<<gblk(Nv), TB, 0, stream>>>(x_var, in_w + 0*16384, in_b + 0*128, T0,  Nv, nullptr, nullptr);
    gemm128_kernel<0,1,false><<<gblk(Nv), TB, 0, stream>>>(T0,    in_w + 1*16384, in_b + 1*128, KT,  Nv, nullptr, nullptr);
    gemm128_kernel<0,0,false><<<gblk(Nv), TB, 0, stream>>>(KT,    in_w + 2*16384, in_b + 2*128, X0a, Nv, nullptr, nullptr);
    gemm128_kernel<0,1,false><<<gblk(Nc), TB, 0, stream>>>(x_con, in_w + 3*16384, in_b + 3*128, T0,  Nc, nullptr, nullptr);
    gemm128_kernel<0,1,false><<<gblk(Nc), TB, 0, stream>>>(T0,    in_w + 4*16384, in_b + 4*128, KT,  Nc, nullptr, nullptr);
    gemm128_kernel<0,0,false><<<gblk(Nc), TB, 0, stream>>>(KT,    in_w + 5*16384, in_b + 5*128, X1a, Nc, nullptr, nullptr);

    float* X0 = X0a; float* X0n = X0b;
    float* X1 = X1a; float* X1n = X1b;
    const dim3 EB((unsigned)((E * 4 + 255) / 256)), ET(256);
    const dim3 SB((unsigned)((E * HC + 255) / 256));

    for (int l = 0; l < L; ++l) {
        const size_t wl0 = (size_t)(l * 2 + 0) * 16384, wl1 = (size_t)(l * 2 + 1) * 16384;
        const size_t bl0 = (size_t)(l * 2 + 0) * 128,   bl1 = (size_t)(l * 2 + 1) * 128;
        const size_t rl0 = (size_t)(l * 2 + 0) * 4096,  rl1 = (size_t)(l * 2 + 1) * 4096;

        // ===== edge type 0: variable -> constraint (dst = constraint) =====
        gemm128_kernel<0,0,false><<<gblk(Nc), TB, 0, stream>>>(X1, q_w + wl1, q_b + bl1, Q,  Nc, nullptr, nullptr);
        gemm128_kernel<0,0,false><<<gblk(Nv), TB, 0, stream>>>(X0, k_w + wl0, k_b + bl0, T0, Nv, nullptr, nullptr);
        rel_transform_kernel<<<gblk(Nv), TB, 0, stream>>>(T0, a_rel + rl0, KT, Nv);
        gemm128_kernel<0,0,false><<<gblk(Nv), TB, 0, stream>>>(X0, v_w + wl0, v_b + bl0, T0, Nv, nullptr, nullptr);
        rel_transform_kernel<<<gblk(Nv), TB, 0, stream>>>(T0, m_rel + rl0, VT, Nv);
        FILL(amax, -INFINITY, Nc * 4);
        FILL(den, 0.0f, Nc * 4);
        FILL(AGGC, 0.0f, Nc * HC);
        attn_score_kernel<<<EB, ET, 0, stream>>>(evc, evc + E, Q, KT, p_rel + (l * 2 + 0) * 4, alpha, amax, E);
        attn_norm_kernel<<<EB, ET, 0, stream>>>(evc + E, alpha, amax, den, E);
        attn_scatter_kernel<<<SB, ET, 0, stream>>>(evc, evc + E, VT, alpha, den, AGGC, E);

        // ===== edge type 1: constraint -> variable (dst = variable) =====
        gemm128_kernel<0,0,false><<<gblk(Nv), TB, 0, stream>>>(X0, q_w + wl0, q_b + bl0, Q,  Nv, nullptr, nullptr);
        gemm128_kernel<0,0,false><<<gblk(Nc), TB, 0, stream>>>(X1, k_w + wl1, k_b + bl1, T0, Nc, nullptr, nullptr);
        rel_transform_kernel<<<gblk(Nc), TB, 0, stream>>>(T0, a_rel + rl1, KT, Nc);
        gemm128_kernel<0,0,false><<<gblk(Nc), TB, 0, stream>>>(X1, v_w + wl1, v_b + bl1, T0, Nc, nullptr, nullptr);
        rel_transform_kernel<<<gblk(Nc), TB, 0, stream>>>(T0, m_rel + rl1, VT, Nc);
        FILL(amax, -INFINITY, Nv * 4);
        FILL(den, 0.0f, Nv * 4);
        FILL(AGGV, 0.0f, Nv * HC);
        attn_score_kernel<<<EB, ET, 0, stream>>>(ecv, ecv + E, Q, KT, p_rel + (l * 2 + 1) * 4, alpha, amax, E);
        attn_norm_kernel<<<EB, ET, 0, stream>>>(ecv + E, alpha, amax, den, E);
        attn_scatter_kernel<<<SB, ET, 0, stream>>>(ecv, ecv + E, VT, alpha, den, AGGV, E);

        // ===== gated update: new = g * (gelu(agg)@a_w + a_b) + (1-g) * x =====
        gemm128_kernel<1,0,true><<<gblk(Nv), TB, 0, stream>>>(AGGV, a_w + wl0, a_b + bl0, X0n, Nv, X0, skip + (l * 2 + 0));
        gemm128_kernel<1,0,true><<<gblk(Nc), TB, 0, stream>>>(AGGC, a_w + wl1, a_b + bl1, X1n, Nc, X1, skip + (l * 2 + 1));
        float* t;
        t = X0; X0 = X0n; X0n = t;
        t = X1; X1 = X1n; X1n = t;
    }

    // ---- output head on variable nodes ----
    gemm128_kernel<0,1,false><<<gblk(Nv), TB, 0, stream>>>(X0, out_w0, out_b0, T0, Nv, nullptr, nullptr);
    gemm128_kernel<0,1,false><<<gblk(Nv), TB, 0, stream>>>(T0, out_w1, out_b1, KT, Nv, nullptr, nullptr);
    FILL(pooled, 0.0f, G * 3);
    head_pool_kernel<<<dim3((unsigned)((Nv + 255) / 256)), dim3(256), 0, stream>>>(KT, out_w2, out_b2, batch, pooled, Nv, G);
    finalize_kernel<<<dim3((unsigned)((G * 2 + 255) / 256)), dim3(256), 0, stream>>>(pooled, (float*)d_out, G);
    #undef FILL
}
